// AMIPRouter_13529146982401
// MI455X (gfx1250) — compile-verified, matrix operand order checked by reading, not run on hardware
//
#include <hip/hip_runtime.h>

typedef __attribute__((ext_vector_type(16))) _Float16 v16h;
typedef __attribute__((ext_vector_type(8)))  float    v8f;

#define SS 2048
#define DD 1024
#define MM 512
#define KK 8
#define D4 256
#define D2 2048
#define MAXW 10    // reference W = 2R = 10 neighbor rows
#define XP 1040    // 1024 + 16 halves pitch (2080 B rows: 32B-aligned, bank-staggered)
#define HP 272     // 256 + 16 halves pitch (544 B rows)
#define TB 4       // tokens per block: each weight fragment feeds 4 WMMAs

__global__ void zero_f32(float* __restrict__ p, size_t n) {
    size_t i = (size_t)blockIdx.x * blockDim.x + threadIdx.x;
    size_t stride = (size_t)gridDim.x * blockDim.x;
    for (; i < n; i += stride) p[i] = 0.f;
}

// W1[k][kd][n] (8,2048,256) f32 -> w1t[(k*256+n)*2048 + kd] f16
__global__ void conv_w1(const float* __restrict__ W1, _Float16* __restrict__ w1t) {
    size_t total = (size_t)KK * D2 * D4;
    size_t i = (size_t)blockIdx.x * blockDim.x + threadIdx.x;
    size_t stride = (size_t)gridDim.x * blockDim.x;
    for (; i < total; i += stride) {
        int n  = (int)(i & (D4 - 1));
        int kd = (int)((i >> 8) & (D2 - 1));
        int k  = (int)(i >> 19);
        w1t[((size_t)(k * D4 + n)) * D2 + kd] = (_Float16)W1[i];
    }
}

// W2[k][kd][n] (8,256,1024) f32 -> w2t[(k*1024+n)*256 + kd] f16
__global__ void conv_w2(const float* __restrict__ W2, _Float16* __restrict__ w2t) {
    size_t total = (size_t)KK * D4 * DD;
    size_t i = (size_t)blockIdx.x * blockDim.x + threadIdx.x;
    size_t stride = (size_t)gridDim.x * blockDim.x;
    for (; i < total; i += stride) {
        int n  = (int)(i & (DD - 1));
        int kd = (int)((i >> 10) & (D4 - 1));
        int k  = (int)(i >> 18);
        w2t[((size_t)(k * DD + n)) * D4 + kd] = (_Float16)W2[i];
    }
}

__global__ void __launch_bounds__(256)
amip_main(const float* __restrict__ hL,
          const int* __restrict__ maskIdx,
          const int* __restrict__ rptr,
          const float* __restrict__ Wr,
          const float* __restrict__ br,
          const _Float16* __restrict__ w1t,
          const float* __restrict__ b1,
          const _Float16* __restrict__ w2t,
          const float* __restrict__ b2,
          float* __restrict__ out)
{
    // ~115 KB static LDS: within gfx1250's 320 KB/workgroup budget, 2 blocks/WGP
    __shared__ __attribute__((aligned(32))) _Float16 xa[TB][MAXW * XP];   // anchor rows
    __shared__ __attribute__((aligned(32))) _Float16 zx[XP];              // shared zero row (x)
    __shared__ __attribute__((aligned(32))) _Float16 hm[TB][XP];          // broadcast h_mask half
    __shared__ __attribute__((aligned(32))) _Float16 hid[TB][MAXW * HP];  // hidden rows
    __shared__ __attribute__((aligned(32))) _Float16 zh[HP];              // shared zero row (hid)
    __shared__ float s_logit[TB][KK];
    __shared__ float s_w[TB][KK];
    __shared__ float s_score[TB][16];
    __shared__ float s_cw[TB][16];
    __shared__ int   s_valid[TB][16];

    const int tid  = threadIdx.x;
    const int wv   = tid >> 5;
    const int lane = tid & 31;
    int R = rptr[0];
    int W = 2 * R; if (W > MAXW) W = MAXW;

    int aIdx[TB], bIdx[TB];
    #pragma unroll
    for (int u = 0; u < TB; ++u) {
        int idx = blockIdx.x * TB + u;
        bIdx[u] = idx / MM;
        aIdx[u] = maskIdx[idx];
    }

    // ---------------- build tiles, routing logits ----------------
    const int k0 = tid * 4;
    // shared zero rows
    {
        zx[k0 + 0] = (_Float16)0.f; zx[k0 + 1] = (_Float16)0.f;
        zx[k0 + 2] = (_Float16)0.f; zx[k0 + 3] = (_Float16)0.f;
        if (tid < 64) {
            zh[k0 + 0] = (_Float16)0.f; zh[k0 + 1] = (_Float16)0.f;
            zh[k0 + 2] = (_Float16)0.f; zh[k0 + 3] = (_Float16)0.f;
        }
    }
    for (int u = 0; u < TB; ++u) {
        const float* hrow = hL + ((size_t)bIdx[u] * SS + aIdx[u]) * DD;
        // h_mask -> f16
        {
            float4 v = *(const float4*)(hrow + k0);
            hm[u][k0 + 0] = (_Float16)v.x; hm[u][k0 + 1] = (_Float16)v.y;
            hm[u][k0 + 2] = (_Float16)v.z; hm[u][k0 + 3] = (_Float16)v.w;
        }
        // anchor rows (clipped neighbor index)
        for (int r = 0; r < W; ++r) {
            int off = (r < R) ? (r - R) : (r - R + 1);
            int t = aIdx[u] + off;
            int tc = t < 0 ? 0 : (t >= SS ? SS - 1 : t);
            float4 v = *(const float4*)(hL + ((size_t)bIdx[u] * SS + tc) * DD + k0);
            _Float16* row = &xa[u][r * XP];
            row[k0 + 0] = (_Float16)v.x; row[k0 + 1] = (_Float16)v.y;
            row[k0 + 2] = (_Float16)v.z; row[k0 + 3] = (_Float16)v.w;
        }
        // routing logit for expert wv
        {
            float s = 0.f;
            for (int d = lane; d < DD; d += 32) s += hrow[d] * Wr[d * KK + wv];
            s += __shfl_xor(s, 16); s += __shfl_xor(s, 8); s += __shfl_xor(s, 4);
            s += __shfl_xor(s, 2);  s += __shfl_xor(s, 1);
            if (lane == 0) s_logit[u][wv] = s + br[wv];
        }
    }
    // valid flags: in-range AND not a masked position (binary search in sorted mask list)
    if (tid < 16 * TB) {
        int u = tid >> 4, r = tid & 15;
        int v = 0;
        if (r < W) {
            int off = (r < R) ? (r - R) : (r - R + 1);
            int t = aIdx[u] + off;
            if (t >= 0 && t < SS) {
                const int* mi = maskIdx + bIdx[u] * MM;
                int lo = 0, hi2 = MM;
                while (lo < hi2) { int mid = (lo + hi2) >> 1; if (mi[mid] < t) lo = mid + 1; else hi2 = mid; }
                int found = (lo < MM) && (mi[lo] == t);
                v = !found;
            }
        }
        s_valid[u][r] = v;
    }
    __syncthreads();
    // relevance scores (dot from f16 LDS); rows >= W never read
    for (int u = 0; u < TB; ++u) {
        for (int r = wv; r < 16; r += 8) {
            float s = 0.f;
            if (r < W) {
                const _Float16* row = &xa[u][r * XP];
                for (int k = lane; k < DD; k += 32) s += (float)row[k] * (float)hm[u][k];
            }
            s += __shfl_xor(s, 16); s += __shfl_xor(s, 8); s += __shfl_xor(s, 4);
            s += __shfl_xor(s, 2);  s += __shfl_xor(s, 1);
            if (lane == 0) s_score[u][r] = s * 0.03125f;  // 1/sqrt(1024)
        }
    }
    __syncthreads();
    if (tid < TB) {  // expert softmax, token u = tid
        int u = tid;
        float mx = -3.4e38f;
        for (int k = 0; k < KK; ++k) mx = fmaxf(mx, s_logit[u][k]);
        float sum = 0.f, e[KK];
        for (int k = 0; k < KK; ++k) { e[k] = __expf(s_logit[u][k] - mx); sum += e[k]; }
        for (int k = 0; k < KK; ++k) s_w[u][k] = e[k] / sum;
    }
    if (tid >= 8 && tid < 8 + TB) {  // combine softmax, token u = tid-8
        int u = tid - 8;
        float sc[16], e[16];
        for (int r = 0; r < 16; ++r) sc[r] = s_valid[u][r] ? s_score[u][r] : -1e30f;
        float mx = -3.4e38f;
        for (int r = 0; r < 16; ++r) mx = fmaxf(mx, sc[r]);
        float sum = 0.f;
        for (int r = 0; r < 16; ++r) { e[r] = __expf(sc[r] - mx); sum += e[r]; }
        for (int r = 0; r < 16; ++r) s_cw[u][r] = s_valid[u][r] ? e[r] / sum : 0.f;
    }
    __syncthreads();

    // ---------------- WMMA main loop: each B fragment feeds TB tokens ----------------
    const int col    = lane & 15;           // A: M row ; B: N column
    const int hi     = lane >> 4;
    const int kshift = hi << 4;
    const _Float16* xaA[TB];
    const _Float16* hidA[TB];
    #pragma unroll
    for (int u = 0; u < TB; ++u) {
        xaA[u]  = (col < W) ? &xa[u][col * XP]  : &zx[0];
        hidA[u] = (col < W) ? &hid[u][col * HP] : &zh[0];
    }

    float cwv[TB][8], cws[TB];
    #pragma unroll
    for (int u = 0; u < TB; ++u) {
        cws[u] = 0.f;
        #pragma unroll
        for (int j = 0; j < 8; ++j) { cwv[u][j] = s_cw[u][j + 8 * hi]; cws[u] += cwv[u][j]; }
    }

    const v8f zf = {0.f, 0.f, 0.f, 0.f, 0.f, 0.f, 0.f, 0.f};
    float dacc[TB][8];
    #pragma unroll
    for (int u = 0; u < TB; ++u)
        #pragma unroll
        for (int t = 0; t < 8; ++t) dacc[u][t] = 0.f;

    for (int i = 0; i < KK; ++i) {
        // ---- GEMM1: hid = gelu(x @ W1[i] + b1[i]); wave owns 2 N-tiles ----
        #pragma unroll
        for (int t = 0; t < 2; ++t) {
            int n0 = (wv * 2 + t) * 16;
            const _Float16* bp = w1t + ((size_t)(i * D4 + n0 + col)) * D2 + kshift;
            v8f c[TB];
            #pragma unroll
            for (int u = 0; u < TB; ++u) c[u] = zf;
            #pragma unroll 2
            for (int kt = 0; kt < DD / 32; ++kt) {      // anchor half of K
                int ko = kt * 32;
                v16h bf = *(const v16h*)(bp + ko);
                #pragma unroll
                for (int u = 0; u < TB; ++u) {
                    v16h af = *(const v16h*)(xaA[u] + ko + kshift);
                    c[u] = __builtin_amdgcn_wmma_f32_16x16x32_f16(false, af, false, bf, (short)0, c[u], false, false);
                }
            }
            #pragma unroll 2
            for (int kt = 0; kt < DD / 32; ++kt) {      // broadcast h_mask half of K
                int ko = kt * 32;
                v16h bf = *(const v16h*)(bp + DD + ko);
                #pragma unroll
                for (int u = 0; u < TB; ++u) {
                    v16h af = *(const v16h*)(&hm[u][ko + kshift]);
                    c[u] = __builtin_amdgcn_wmma_f32_16x16x32_f16(false, af, false, bf, (short)0, c[u], false, false);
                }
            }
            float bias = b1[i * D4 + n0 + col];
            #pragma unroll
            for (int j = 0; j < 8; ++j) {
                int row = j + 8 * hi;
                if (row < W) {
                    #pragma unroll
                    for (int u = 0; u < TB; ++u) {
                        float x = c[u][j] + bias;
                        float g = 0.5f * x * (1.f + erff(x * 0.70710678f));  // exact gelu
                        hid[u][row * HP + n0 + col] = (_Float16)g;
                    }
                }
            }
        }
        __syncthreads();
        float wi[TB];
        #pragma unroll
        for (int u = 0; u < TB; ++u) wi[u] = s_w[u][i];
        // ---- GEMM2 + early cw-fold: dacc += w_i * (cw . (hid@W2 + b2)) ----
        #pragma unroll
        for (int t = 0; t < 8; ++t) {
            int n0 = wv * 128 + t * 16;
            const _Float16* bp = w2t + ((size_t)(i * DD + n0 + col)) * D4 + kshift;
            v8f c[TB];
            #pragma unroll
            for (int u = 0; u < TB; ++u) c[u] = zf;
            #pragma unroll
            for (int kt = 0; kt < D4 / 32; ++kt) {
                int ko = kt * 32;
                v16h bf = *(const v16h*)(bp + ko);
                #pragma unroll
                for (int u = 0; u < TB; ++u) {
                    v16h af = *(const v16h*)(hidA[u] + ko + kshift);
                    c[u] = __builtin_amdgcn_wmma_f32_16x16x32_f16(false, af, false, bf, (short)0, c[u], false, false);
                }
            }
            float bias = b2[i * DD + n0 + col];
            #pragma unroll
            for (int u = 0; u < TB; ++u) {
                float s = 0.f;
                #pragma unroll
                for (int j = 0; j < 8; ++j) s += cwv[u][j] * c[u][j];
                dacc[u][t] += wi[u] * (s + bias * cws[u]);
            }
        }
        __syncthreads();
    }

    // ---------------- cross-lane-half reduce, store delta rows ----------------
    #pragma unroll
    for (int u = 0; u < TB; ++u) {
        float* orow = out + ((size_t)bIdx[u] * SS + aIdx[u]) * DD;
        #pragma unroll
        for (int t = 0; t < 8; ++t) {
            float s = dacc[u][t];
            s += __shfl_xor(s, 16);   // add rows 8..15 partial from upper lane half
            if (hi == 0) orow[wv * 128 + t * 16 + col] = s;
        }
    }
}

extern "C" void kernel_launch(void* const* d_in, const int* in_sizes, int n_in,
                              void* d_out, int out_size, void* d_ws, size_t ws_size,
                              hipStream_t stream) {
    const float* hL   = (const float*)d_in[0];
    const int*   mIdx = (const int*)  d_in[1];
    // d_in[2] = unmasked_indices (unused; validity derived from sorted mask_indices)
    const int*   rptr = (const int*)  d_in[3];
    const float* Wr   = (const float*)d_in[4];
    const float* br   = (const float*)d_in[5];
    const float* W1   = (const float*)d_in[6];
    const float* b1   = (const float*)d_in[7];
    const float* W2   = (const float*)d_in[8];
    const float* b2   = (const float*)d_in[9];
    float* out = (float*)d_out;

    _Float16* w1t = (_Float16*)d_ws;                                          // 8 MiB
    _Float16* w2t = (_Float16*)((char*)d_ws + (size_t)KK * D2 * D4 * 2);      // +4 MiB

    zero_f32<<<4096, 256, 0, stream>>>(out, (size_t)out_size);
    conv_w1<<<4096, 256, 0, stream>>>(W1, w1t);
    conv_w2<<<2048, 256, 0, stream>>>(W2, w2t);

    int ntok   = in_sizes[1];       // B*M masked tokens
    int nblock = ntok / TB;         // TB tokens per block
    amip_main<<<nblock, 256, 0, stream>>>(hL, mIdx, rptr, Wr, br, w1t, b1, w2t, b2, out);
}
